// GraphAttentionNetwork_12489764897018
// MI455X (gfx1250) — matrix operand order repaired and checked by reading, not compiled, not run on hardware
//
#include <hip/hip_runtime.h>

// GAT forward for MI455X (gfx1250, wave32, WMMA).
// B=8, N=1024, F=1024, H=16, D=64.
// Pipeline:
//   k_cvt_x : x f32 [B,N,F] -> xb bf16 [B,N,F]
//   k_cvt_w : W f32 [H,F,D] -> wtb bf16 [H,D,F] (transposed)
//   k_gemm  : Wh = x @ W via v_wmma_f32_16x16x32_bf16 (register double-buffered);
//             emits whT bf16 [B,H,D,N] (packed b128 stores) and
//             f1/f2 = Wh . a1/a2 (in-register cross-lane reduction).
//   k_attn  : per (b,h,16-row tile):
//             pass A: masked leaky-relu scores kept in regs, row max via shfl
//             pass B: ONE exp per element -> row sum + P bf16 tile in LDS
//             pass C: O = P @ Wh via WMMA, A-fragment from LDS, double-buffered
//             store y = O / rowsum.
// Workspace: ~35 MiB.

typedef __attribute__((ext_vector_type(16))) __bf16 v16bf;
typedef __attribute__((ext_vector_type(8)))  __bf16 v8bf;
typedef __attribute__((ext_vector_type(8)))  float  v8f;

#define GAT_B 8
#define GAT_N 1024
#define GAT_F 1024
#define GAT_H 16
#define GAT_D 64
#define NEG_BIG_F (-9.0e15f)

// Load a 16-bit WMMA fragment: 8 contiguous elems at p, 8 at p+hioff.
// A-fragment (16x32): hioff=16.  B-fragment (32x16, K-contiguous): hioff=16.
__device__ __forceinline__ v16bf ld_frag(const __bf16* p, int hioff) {
    v8bf lo = *(const v8bf*)p;
    v8bf hi = *(const v8bf*)(p + hioff);
    v16bf r;
    #pragma unroll
    for (int e = 0; e < 8; ++e) { r[e] = lo[e]; r[e + 8] = hi[e]; }
    return r;
}

// ---------------- conversion kernels ----------------

__global__ void k_cvt_x(const float* __restrict__ x, __bf16* __restrict__ xb, int n) {
    int i = blockIdx.x * blockDim.x + threadIdx.x;
    if (i < n) xb[i] = (__bf16)x[i];
}

// W [H,F,D] -> wtb [H,D,F]
__global__ void k_cvt_w(const float* __restrict__ W, __bf16* __restrict__ wtb) {
    int i = blockIdx.x * blockDim.x + threadIdx.x;
    const int total = GAT_H * GAT_F * GAT_D;
    if (i >= total) return;
    int d = i & (GAT_D - 1);
    int f = (i >> 6) & (GAT_F - 1);
    int h = i >> 16;
    wtb[((size_t)h * GAT_D + d) * GAT_F + f] = (__bf16)W[i];
}

// ---------------- fused GEMM: Wh = x @ W, + whT bf16, + f1/f2 ----------------
// one wave computes a 16x64 tile of Wh for one (b,h)
__global__ void k_gemm(const __bf16* __restrict__ xb,   // [B,N,F]
                       const __bf16* __restrict__ wtb,  // [H,D,F]
                       const float* __restrict__ a1,    // [H,D]
                       const float* __restrict__ a2,    // [H,D]
                       __bf16* __restrict__ whT,        // [B*H, D, N]
                       float* __restrict__ f1,          // [B*H, N]
                       float* __restrict__ f2)          // [B*H, N]
{
    int wave = (blockIdx.x * blockDim.x + threadIdx.x) >> 5;
    int lane = threadIdx.x & 31;
    int rowTile = wave & 63;            // N/16 tiles
    int bh = wave >> 6;                 // b*H + h
    int h  = bh & (GAT_H - 1);
    int b  = bh >> 4;
    int lm = lane & 15;
    int hs = (lane >> 4) & 1;

    const __bf16* arow = xb + ((size_t)b * GAT_N + rowTile * 16 + lm) * GAT_F + hs * 8;
    const __bf16* bbase = wtb + ((size_t)h * GAT_D + lm) * GAT_F + hs * 8;

    v8f acc[4] = {};
    // software pipeline: prefetch fragments for next k while WMMAs consume current
    v16bf af = ld_frag(arow, 16);
    v16bf bf[4];
    #pragma unroll
    for (int nt = 0; nt < 4; ++nt) bf[nt] = ld_frag(bbase + (size_t)nt * 16 * GAT_F, 16);

    for (int k0 = 0; k0 < GAT_F; k0 += 32) {
        int kn = (k0 + 32) & (GAT_F - 1);          // wraps to 0 on last iter (harmless)
        v16bf afn = ld_frag(arow + kn, 16);
        v16bf bfn[4];
        #pragma unroll
        for (int nt = 0; nt < 4; ++nt)
            bfn[nt] = ld_frag(bbase + (size_t)nt * 16 * GAT_F + kn, 16);
        #pragma unroll
        for (int nt = 0; nt < 4; ++nt)
            acc[nt] = __builtin_amdgcn_wmma_f32_16x16x32_bf16(
                false, af, false, bf[nt], (short)0, acc[nt], false, false);
        af = afn;
        #pragma unroll
        for (int nt = 0; nt < 4; ++nt) bf[nt] = bfn[nt];
    }

    // ---- store whT[b,h,d, n] bf16: lane owns column d = nt*16+lm, rows contiguous ----
    __bf16* tp = whT + (size_t)bh * GAT_D * GAT_N + rowTile * 16 + hs * 8;
    #pragma unroll
    for (int nt = 0; nt < 4; ++nt) {
        v8bf pk;
        #pragma unroll
        for (int r = 0; r < 8; ++r) pk[r] = (__bf16)acc[nt][r];
        *(v8bf*)(tp + (size_t)(nt * 16 + lm) * GAT_N) = pk;
    }

    // ---- f1/f2 = Wh . a1/a2 : reduce over d (= across 16 lanes and nt) ----
    float a1v[4], a2v[4];
    #pragma unroll
    for (int nt = 0; nt < 4; ++nt) {
        a1v[nt] = a1[h * GAT_D + nt * 16 + lm];
        a2v[nt] = a2[h * GAT_D + nt * 16 + lm];
    }
    #pragma unroll
    for (int r = 0; r < 8; ++r) {
        float p1 = 0.f, p2 = 0.f;
        #pragma unroll
        for (int nt = 0; nt < 4; ++nt) {
            p1 += acc[nt][r] * a1v[nt];
            p2 += acc[nt][r] * a2v[nt];
        }
        // reduce within the 16-lane half (d dimension); offsets 8..1 stay in-half
        #pragma unroll
        for (int off = 8; off > 0; off >>= 1) {
            p1 += __shfl_xor(p1, off, 32);
            p2 += __shfl_xor(p2, off, 32);
        }
        if (lm == 0) {
            int grow = rowTile * 16 + hs * 8 + r;
            f1[(size_t)bh * GAT_N + grow] = p1;
            f2[(size_t)bh * GAT_N + grow] = p2;
        }
    }
}

// ---------------- fused masked softmax + WMMA aggregation ----------------
// one wave handles one (b,h, 16-row tile); P tile (16x1024 bf16, 32KB) in LDS
__global__ void k_attn(const float* __restrict__ f1,   // [B*H, N]
                       const float* __restrict__ f2,   // [B*H, N]
                       const int*   __restrict__ adj,  // [B, N, N]
                       const __bf16* __restrict__ whT, // [B*H, D, N]
                       float* __restrict__ y)          // [B, N, H*D]
{
    extern __shared__ __bf16 smem[];
    int wave = (blockIdx.x * blockDim.x + threadIdx.x) >> 5;
    int lane = threadIdx.x & 31;
    int rowTile = wave & 63;
    int bh = wave >> 6;
    int h  = bh & (GAT_H - 1);
    int b  = bh >> 4;
    int lm = lane & 15;
    int hs = (lane >> 4) & 1;
    __bf16* pl = smem + (size_t)(threadIdx.x >> 5) * 16 * GAT_N;  // 32KB per wave

    const float* f1p  = f1 + (size_t)bh * GAT_N + rowTile * 16;
    const float* f2p  = f2 + (size_t)bh * GAT_N;
    const int*   adjp = adj + ((size_t)b * GAT_N + rowTile * 16) * GAT_N;
    int jbase = lane * 32;                       // each lane owns 32 contiguous j

    // hoist this lane's f2 chunk (invariant across rows)
    float f2v[32];
    {
        const float4* q4 = (const float4*)(f2p + jbase);
        #pragma unroll
        for (int t = 0; t < 8; ++t) {
            float4 fv = q4[t];
            f2v[4*t+0] = fv.x; f2v[4*t+1] = fv.y; f2v[4*t+2] = fv.z; f2v[4*t+3] = fv.w;
        }
    }

    float sv[8] = {};     // row sums for rows hs*8 .. hs*8+7 (store phase)
    #pragma unroll
    for (int r = 0; r < 16; ++r) {
        float f1r = f1p[r];
        const int4* a4 = (const int4*)(adjp + (size_t)r * GAT_N + jbase);
        // pass A: scores in registers, row max (no exp)
        float sc[32];
        float lmax = -3.0e38f;
        #pragma unroll
        for (int t = 0; t < 8; ++t) {
            int4 av = a4[t];
            float e0 = f1r + f2v[4*t+0]; e0 = e0 > 0.f ? e0 : 0.2f * e0;
            float e1 = f1r + f2v[4*t+1]; e1 = e1 > 0.f ? e1 : 0.2f * e1;
            float e2 = f1r + f2v[4*t+2]; e2 = e2 > 0.f ? e2 : 0.2f * e2;
            float e3 = f1r + f2v[4*t+3]; e3 = e3 > 0.f ? e3 : 0.2f * e3;
            sc[4*t+0] = av.x > 0 ? e0 : NEG_BIG_F;
            sc[4*t+1] = av.y > 0 ? e1 : NEG_BIG_F;
            sc[4*t+2] = av.z > 0 ? e2 : NEG_BIG_F;
            sc[4*t+3] = av.w > 0 ? e3 : NEG_BIG_F;
            lmax = fmaxf(lmax, fmaxf(fmaxf(sc[4*t+0], sc[4*t+1]),
                                     fmaxf(sc[4*t+2], sc[4*t+3])));
        }
        #pragma unroll
        for (int off = 16; off > 0; off >>= 1)
            lmax = fmaxf(lmax, __shfl_xor(lmax, off, 32));
        // pass B: ONE exp per element; row sum + P bf16 into LDS
        float lsum = 0.f;
        #pragma unroll
        for (int q = 0; q < 4; ++q) {
            v8bf pk;
            #pragma unroll
            for (int t = 0; t < 8; ++t) {
                float p = __expf(sc[q*8+t] - lmax);
                lsum += p;
                pk[t] = (__bf16)p;
            }
            *(v8bf*)(pl + (size_t)r * GAT_N + jbase + q * 8) = pk;
        }
        #pragma unroll
        for (int off = 16; off > 0; off >>= 1)
            lsum += __shfl_xor(lsum, off, 32);
        if ((r >> 3) == hs) sv[r & 7] = lsum;
    }

    // pass C: O = P @ Wh via WMMA; A-fragment from LDS, double-buffered
    const __bf16* plds = pl + (size_t)lm * GAT_N + hs * 8;
    const __bf16* wb = whT + ((size_t)bh * GAT_D + lm) * GAT_N + hs * 8;
    v8f acc[4] = {};
    v16bf pf = ld_frag(plds, 16);
    v16bf bf[4];
    #pragma unroll
    for (int nt = 0; nt < 4; ++nt) bf[nt] = ld_frag(wb + (size_t)nt * 16 * GAT_N, 16);

    for (int j0 = 0; j0 < GAT_N; j0 += 32) {
        int jn = (j0 + 32) & (GAT_N - 1);
        v16bf pfn = ld_frag(plds + jn, 16);
        v16bf bfn[4];
        #pragma unroll
        for (int nt = 0; nt < 4; ++nt)
            bfn[nt] = ld_frag(wb + (size_t)nt * 16 * GAT_N + jn, 16);
        #pragma unroll
        for (int nt = 0; nt < 4; ++nt)
            acc[nt] = __builtin_amdgcn_wmma_f32_16x16x32_bf16(
                false, pf, false, bf[nt], (short)0, acc[nt], false, false);
        pf = pfn;
        #pragma unroll
        for (int nt = 0; nt < 4; ++nt) bf[nt] = bfn[nt];
    }

    // ---- store y[b, row, h*D + d] = O / rowsum ----
    #pragma unroll
    for (int nt = 0; nt < 4; ++nt)
        #pragma unroll
        for (int r = 0; r < 8; ++r) {
            int m = r + hs * 8;
            int grow = rowTile * 16 + m;
            float val = acc[nt][r] / sv[r];
            y[((size_t)b * GAT_N + grow) * (GAT_H * GAT_D) + h * GAT_D
              + nt * 16 + lm] = val;
        }
}

// ---------------- launch ----------------
extern "C" void kernel_launch(void* const* d_in, const int* in_sizes, int n_in,
                              void* d_out, int out_size, void* d_ws, size_t ws_size,
                              hipStream_t stream) {
    const float* x   = (const float*)d_in[0];  // [B,N,F]
    const int*   adj = (const int*)  d_in[1];  // [B,N,N]
    const float* W   = (const float*)d_in[2];  // [H,F,D]
    const float* a1  = (const float*)d_in[3];  // [H,D]
    const float* a2  = (const float*)d_in[4];  // [H,D]
    float* y = (float*)d_out;                  // [B,N,H*D]

    char* ws = (char*)d_ws;
    const size_t MiB = 1024 * 1024;
    __bf16* xb  = (__bf16*)(ws);               // 16 MiB
    __bf16* wtb = (__bf16*)(ws + 16 * MiB);    //  2 MiB
    __bf16* whT = (__bf16*)(ws + 18 * MiB);    // 16 MiB
    float*  f1  = (float*) (ws + 34 * MiB);    // 0.5 MiB
    float*  f2  = (float*) (ws + 34 * MiB + 512 * 1024); // 0.5 MiB

    const int nx = GAT_B * GAT_N * GAT_F;      // 8M
    k_cvt_x<<<(nx + 255) / 256, 256, 0, stream>>>(x, xb, nx);

    const int nw = GAT_H * GAT_F * GAT_D;      // 1M
    k_cvt_w<<<(nw + 255) / 256, 256, 0, stream>>>(W, wtb);

    // 8192 waves, 8 waves (256 threads) per block
    k_gemm<<<1024, 256, 0, stream>>>(xb, wtb, a1, a2, whT, f1, f2);

    // 8192 waves, 4 waves (128 threads) per block, 4 x 32KB LDS for P tiles
    k_attn<<<2048, 128, 4 * 16 * GAT_N * sizeof(__bf16), stream>>>(f1, f2, adj, whT, y);
}
